// DHGAT_NET_45569603011139
// MI455X (gfx1250) — matrix operand order repaired and checked by reading, not compiled, not run on hardware
//
#include <hip/hip_runtime.h>
#include <math.h>

#define NN  40000
#define EE  640000
#define TIN 128   // inner dim of every WMMA GEMM in this net (IN == HID == 128)
#define LDX 136   // padded LDS row stride in halfs (272B = 17*16B: aligned + bank-rotating)

typedef __attribute__((ext_vector_type(16))) _Float16 v16h;
typedef __attribute__((ext_vector_type(8)))  _Float16 v8h;
typedef __attribute__((ext_vector_type(8)))  float    v8f;

union FragA { v16h v; v8h g[2]; _Float16 h[16]; };
union FragC { v8f  v; float f[8]; };

// ---------------------------------------------------------------- utilities
__global__ void k_zero(float* p, int n) {
  int i = blockIdx.x * blockDim.x + threadIdx.x;
  int s = gridDim.x * blockDim.x;
  for (; i < n; i += s) p[i] = 0.0f;
}

__global__ void k_relu(float* p, int n) {
  int i = blockIdx.x * blockDim.x + threadIdx.x;
  int s = gridDim.x * blockDim.x;
  for (; i < n; i += s) p[i] = fmaxf(p[i], 0.0f);
}

// ------------------------------------------------------------- WMMA GEMM
// H[N,fout] = X[N,128] @ W[128,fout] (+bias)(+relu), f16 operands / f32 acc.
// One wave -> one 16x16 output tile; block covers 16 rows x fout cols.
// LDS: sX[16][128] row-major (padded), sWt[fout][128] = W transposed (padded),
// so every A/B fragment is two contiguous 16B chunks -> ds_load_b128.
__global__ __launch_bounds__(256)
void k_gemm(const float* __restrict__ X, const float* __restrict__ W,
            const float* __restrict__ bias, float* __restrict__ H,
            int fout, int fsh /* log2(fout) */, int relu) {
  extern __shared__ _Float16 smem[];
  _Float16* sX  = smem;             // [16][LDX]
  _Float16* sWt = smem + 16 * LDX;  // [fout][LDX]
  const int m0 = blockIdx.x * 16;

  for (int idx = threadIdx.x; idx < 16 * TIN; idx += blockDim.x) {
    int r = idx >> 7, c = idx & 127;
    sX[r * LDX + c] = (_Float16)X[(size_t)(m0 + r) * TIN + c];
  }
  // coalesced global read of W[k][c]; transposed (bank-rotated) LDS write
  for (int idx = threadIdx.x; idx < TIN * fout; idx += blockDim.x) {
    int k = idx >> fsh, c = idx & (fout - 1);
    sWt[c * LDX + k] = (_Float16)W[idx];
  }
  __syncthreads();

  const int lane = threadIdx.x & 31;
  const int wid  = threadIdx.x >> 5;
  const int r16  = lane & 15;
  const int kh   = lane >> 4;
  const int col  = wid * 16 + r16;

  const _Float16* ax = sX  + r16 * LDX;   // A row for this lane
  const _Float16* bx = sWt + col * LDX;   // B column for this lane

  FragC acc = {};
#pragma unroll
  for (int kk = 0; kk < TIN; kk += 32) {
    FragA A, B;
    // A.h[i] (i<8)  = X[row, kk + 8*kh + i];  A.h[8+i] = X[row, kk+16+8*kh+i]
    A.g[0] = *(const v8h*)(ax + kk + 8 * kh);
    A.g[1] = *(const v8h*)(ax + kk + 16 + 8 * kh);
    // B.h[i] (i<8) = W[kk + 8*kh + i, col];  B.h[8+i] = W[kk+16+8*kh+i, col]
    B.g[0] = *(const v8h*)(bx + kk + 8 * kh);
    B.g[1] = *(const v8h*)(bx + kk + 16 + 8 * kh);
    acc.v = __builtin_amdgcn_wmma_f32_16x16x32_f16(false, A.v, false, B.v,
                                                   (short)0, acc.v, false, false);
  }
  const float b = bias ? bias[col] : 0.0f;
  // D layout: VGPR r -> M = r + 8*kh, N = lane%16
#pragma unroll
  for (int r = 0; r < 8; ++r) {
    float v = acc.f[r] + b;
    if (relu) v = fmaxf(v, 0.0f);
    H[(size_t)(m0 + r + 8 * kh) * fout + col] = v;
  }
}

// -------------------------------------------------- decision head (T = 2)
__global__ void k_decide(const float* __restrict__ tmp, const float* __restrict__ w2,
                         const float* __restrict__ b2, float* __restrict__ wgt, int n) {
  int i = blockIdx.x * blockDim.x + threadIdx.x;
  if (i >= n) return;
  const float* r = tmp + (size_t)i * 64;
  float z0 = b2[0], z1 = b2[1];
#pragma unroll
  for (int k = 0; k < 64; ++k) {
    float v = r[k];
    z0 += v * w2[2 * k];
    z1 += v * w2[2 * k + 1];
  }
  float m  = fmaxf(z0, z1);
  float e0 = __expf(z0 - m), e1 = __expf(z1 - m);
  float iv = 1.0f / (e0 + e1);
  wgt[2 * i]     = e0 * iv;
  wgt[2 * i + 1] = e1 * iv;
}

// -------------------------------------- per-node attention scalars (wave/node)
__global__ void k_attn(const float* __restrict__ h, const float* __restrict__ a0,
                       const float* __restrict__ a1, float* __restrict__ s0,
                       float* __restrict__ s1, int n, int f) {
  int node = blockIdx.x * (blockDim.x >> 5) + (threadIdx.x >> 5);
  if (node >= n) return;
  int lane = threadIdx.x & 31;
  float p0 = 0.0f, p1 = 0.0f;
  for (int k = lane; k < f; k += 32) {
    float hv = h[(size_t)node * f + k];
    p0 += hv * a0[k];
    p1 += hv * a1[k];
  }
#pragma unroll
  for (int m = 16; m; m >>= 1) {
    p0 += __shfl_xor(p0, m, 32);
    p1 += __shfl_xor(p1, m, 32);
  }
  if (lane == 0) { s0[node] = p0; s1[node] = p1; }
}

// ------------------------------------------ edge pass 1: score + segment max
__device__ __forceinline__ unsigned f2key(float v) {
  unsigned b = __float_as_uint(v);
  return (b & 0x80000000u) ? ~b : (b | 0x80000000u);
}
__device__ __forceinline__ float key2f(unsigned k) {
  return (k & 0x80000000u) ? __uint_as_float(k & 0x7FFFFFFFu) : __uint_as_float(~k);
}

__global__ void k_edge1(const int* __restrict__ src, const int* __restrict__ dst,
                        const float* __restrict__ s0, const float* __restrict__ s1,
                        float* __restrict__ ev, unsigned* __restrict__ mkey, int e) {
  int i = blockIdx.x * blockDim.x + threadIdx.x;
  if (i >= e) return;
  float v = s0[src[i]] + s1[dst[i]];
  v = (v >= 0.0f) ? v : 0.2f * v;     // leaky_relu 0.2
  ev[i] = v;
  atomicMax(mkey + dst[i], f2key(v)); // order-preserving uint key (init 0 == -inf)
}

// ------------------------------------------ edge pass 2: exp + segment sum
__global__ void k_edge2(const int* __restrict__ dst, const unsigned* __restrict__ mkey,
                        float* __restrict__ ssum, float* __restrict__ ev, int e) {
  int i = blockIdx.x * blockDim.x + threadIdx.x;
  if (i >= e) return;
  int d = dst[i];
  float ex = __expf(ev[i] - key2f(mkey[d]));
  ev[i] = ex;
  atomicAdd(ssum + d, ex);
}

// -------------------- edge pass 3: weighted scatter (wave/edge, coalesced)
__global__ __launch_bounds__(256)
void k_scatter(const int* __restrict__ src, const int* __restrict__ dst,
               const float* __restrict__ h, const float* __restrict__ ev,
               const float* __restrict__ ssum, const float* __restrict__ wgt,
               int t, float* __restrict__ out, int e, int f) {
  int edge = blockIdx.x * (blockDim.x >> 5) + (threadIdx.x >> 5);
  if (edge >= e) return;
  int lane = threadIdx.x & 31;
  int s = src[edge], d = dst[edge];
  float alpha = ev[edge] / (ssum[d] + 1e-9f) * wgt[2 * d + t];
  for (int k = lane; k < f; k += 32)
    atomicAdd(out + (size_t)d * f + k, alpha * h[(size_t)s * f + k]);
}

// ---------------------------------------------------------- classifier head
__global__ void k_head(const float* __restrict__ h, const float* __restrict__ w1,
                       const float* __restrict__ b1, const float* __restrict__ w2,
                       const float* __restrict__ b2, float* __restrict__ out, int n) {
  int i = blockIdx.x * blockDim.x + threadIdx.x;
  if (i >= n) return;
  const float* r = h + (size_t)i * 64;
  float t1[32];
#pragma unroll
  for (int j = 0; j < 32; ++j) t1[j] = b1[j];
  for (int k = 0; k < 64; ++k) {
    float v = r[k];
#pragma unroll
    for (int j = 0; j < 32; ++j) t1[j] += v * w1[k * 32 + j];
  }
  float z0 = b2[0], z1 = b2[1];
#pragma unroll
  for (int j = 0; j < 32; ++j) {
    float v = fmaxf(t1[j], 0.0f);
    z0 += v * w2[2 * j];
    z1 += v * w2[2 * j + 1];
  }
  float m  = fmaxf(z0, z1);
  float e0 = __expf(z0 - m), e1 = __expf(z1 - m);
  float iv = 1.0f / (e0 + e1);
  out[2 * i]     = e0 * iv;
  out[2 * i + 1] = e1 * iv;
}

// --------------------------------------------------------------------------
extern "C" void kernel_launch(void* const* d_in, const int* in_sizes, int n_in,
                              void* d_out, int out_size, void* d_ws, size_t ws_size,
                              hipStream_t stream) {
  const float* x    = (const float*)d_in[0];
  const int*   eia  = (const int*)  d_in[1];   // [2,E]: row0=src, row1=dst
  const int*   eib  = (const int*)  d_in[2];
  const float* W1   = (const float*)d_in[3];   // [2,128,128]
  const float* a1   = (const float*)d_in[4];   // [2,2,128]
  const float* d1w1 = (const float*)d_in[5];
  const float* d1b1 = (const float*)d_in[6];
  const float* d1w2 = (const float*)d_in[7];
  const float* d1b2 = (const float*)d_in[8];
  const float* W2   = (const float*)d_in[9];   // [2,128,64]
  const float* a2   = (const float*)d_in[10];  // [2,2,64]
  const float* d2w1 = (const float*)d_in[11];
  const float* d2b1 = (const float*)d_in[12];
  const float* d2w2 = (const float*)d_in[13];
  const float* d2b2 = (const float*)d_in[14];
  const float* f1w  = (const float*)d_in[15];
  const float* f1b  = (const float*)d_in[16];
  const float* f2w  = (const float*)d_in[17];
  const float* f2b  = (const float*)d_in[18];
  float* out = (float*)d_out;

  // workspace carve-up (~65 MB)
  float* ws   = (float*)d_ws;
  float* wgt  = ws;  ws += (size_t)NN * 2;
  float* tmp  = ws;  ws += (size_t)NN * 64;
  float* h    = ws;  ws += (size_t)NN * 128;
  float* s0   = ws;  ws += NN;
  float* s1   = ws;  ws += NN;
  unsigned* mkey = (unsigned*)ws; ws += NN;
  float* ssum = ws;  ws += NN;
  float* ev   = ws;  ws += EE;
  float* x1   = ws;  ws += (size_t)NN * 128;
  float* x2   = ws;  ws += (size_t)NN * 64;

  auto layer = [&](const float* X, int fout, int fsh, const float* W, const float* a,
                   const float* dw1, const float* db1, const float* dw2,
                   const float* db2, float* xout) {
    // decision MLP: relu(X @ dw1 + db1) via WMMA, then softmax over 2 types
    size_t smemD = (size_t)(16 * LDX + 64 * LDX) * sizeof(_Float16);
    k_gemm<<<NN / 16, 128, smemD, stream>>>(X, dw1, db1, tmp, 64, 6, 1);
    k_decide<<<(NN + 255) / 256, 256, 0, stream>>>(tmp, dw2, db2, wgt, NN);
    k_zero<<<2048, 256, 0, stream>>>(xout, NN * fout);

    size_t smemG = (size_t)(16 * LDX + fout * LDX) * sizeof(_Float16);
    for (int t = 0; t < 2; ++t) {
      const int* src = (t == 0 ? eia : eib);
      const int* dst = src + EE;
      k_gemm<<<NN / 16, 32 * (fout / 16), smemG, stream>>>(
          X, W + (size_t)t * TIN * fout, nullptr, h, fout, fsh, 0);
      k_attn<<<(NN + 7) / 8, 256, 0, stream>>>(
          h, a + (size_t)t * 2 * fout, a + (size_t)t * 2 * fout + fout, s0, s1, NN, fout);
      k_zero<<<64, 256, 0, stream>>>((float*)mkey, NN);  // key 0 == -inf
      k_zero<<<64, 256, 0, stream>>>(ssum, NN);
      k_edge1<<<(EE + 255) / 256, 256, 0, stream>>>(src, dst, s0, s1, ev, mkey, EE);
      k_edge2<<<(EE + 255) / 256, 256, 0, stream>>>(dst, mkey, ssum, ev, EE);
      k_scatter<<<(EE + 7) / 8, 256, 0, stream>>>(src, dst, h, ev, ssum, wgt, t,
                                                  xout, EE, fout);
    }
    k_relu<<<2048, 256, 0, stream>>>(xout, NN * fout);
  };

  layer(x,  128, 7, W1, a1, d1w1, d1b1, d1w2, d1b2, x1);
  layer(x1,  64, 6, W2, a2, d2w1, d2b1, d2w2, d2b2, x2);
  k_head<<<(NN + 255) / 256, 256, 0, stream>>>(x2, f1w, f1b, f2w, f2b, out, NN);
}
